// BaseRNN_66271345377472
// MI455X (gfx1250) — matrix operand order repaired and checked by reading, not compiled, not run on hardware
//
#include <hip/hip_runtime.h>
#include <hip/hip_bf16.h>

typedef __bf16 bf16;
typedef __attribute__((ext_vector_type(16))) __bf16 v16bf;
typedef __attribute__((ext_vector_type(8)))  float  v8f;

#define TT   16
#define BB   4
#define CIN  32
#define HID  64
#define HH   64
#define WW   64
#define CHTOT 96            // CIN + HID
#define KTOT 864            // CHTOT * 9
#define KCH  27             // KTOT / 32 k-chunks
#define GOUT 256            // 4 * HID gate channels
#define PIX  (HH*WW)        // 4096

// ---------------------------------------------------------------- prep: x -> bf16
__global__ __launch_bounds__(256) void cvt_x_kernel(const float* __restrict__ x,
                                                    bf16* __restrict__ x16, int n) {
  int i = blockIdx.x * 256 + threadIdx.x;
  if (i < n) x16[i] = (bf16)x[i];
}

// ---------------------------------------------------------------- prep: pack weights
// Packed A-fragment layout: [m_tile(16)][k_chunk(27)][lane(32)][16 bf16]
// ISA 16-bit A 16x32 layout: lane<16 -> M=lane, K_local = (e<8 ? e : e+8)
//                            lane>=16 -> M=lane-16, K_local += 8
__global__ __launch_bounds__(256) void pack_w_kernel(const float* __restrict__ wx,
                                                     const float* __restrict__ wh,
                                                     bf16* __restrict__ wp) {
  int idx  = blockIdx.x * 256 + threadIdx.x;      // < 16*27*32*16 = 221184
  int e    = idx & 15;
  int lane = (idx >> 4) & 31;
  int frag = idx >> 9;
  int kc   = frag % KCH;
  int mt   = frag / KCH;
  int kl   = (e < 8 ? e : e + 8) + ((lane < 16) ? 0 : 8);
  int K    = kc * 32 + kl;                        // 0..863
  int ch   = K / 9;
  int tap  = K - ch * 9;
  int m    = mt * 16 + (lane & 15);
  float v  = (ch < CIN) ? wx[(m * CIN + ch) * 9 + tap]
                        : wh[(m * HID + (ch - CIN)) * 9 + tap];
  wp[idx] = (bf16)v;
}

// ---------------------------------------------------------------- prep: zero h,c
__global__ __launch_bounds__(256) void zero_state_kernel(float* __restrict__ c,
                                                         bf16* __restrict__ h16) {
  int i = blockIdx.x * 256 + threadIdx.x;         // grid covers BB*HID*PIX
  if (i < BB * HID * PIX) { c[i] = 0.f; h16[i] = (bf16)0.f; }
}

// ---------------------------------------------------------------- fused conv (implicit GEMM)
// grid = BB * HH * (WW/32) = 512 workgroups, 256 threads (8 waves).
// Workgroup: 32 pixels (one row segment) x all 256 gate channels.
// Wave: 32 channels x 32 pixels -> 2x2 accumulator grid; K loop of 27 chunks,
// 4 x wmma_f32_16x16x32_bf16 per chunk (A reused over 2 N-tiles, B over 2 M-tiles).
__global__ __launch_bounds__(256) void conv_gates_kernel(
    const bf16* __restrict__ x16,   // [BB,CIN,HH,WW] for this timestep
    const bf16* __restrict__ h16,   // [BB,HID,HH,WW]
    const bf16* __restrict__ wp,    // packed A fragments
    float* __restrict__ gates) {    // [BB,GOUT,HH,WW]
  __shared__ v16bf ldsB[KCH * 64];  // 27 chunks x 2 N-tiles x 32 lanes x 32B = 55296 B
  bf16* ldsBs = (bf16*)ldsB;

  int wg   = blockIdx.x;
  int b    = wg >> 7;               // 128 wgs per batch image
  int rem  = wg & 127;
  int y0   = rem >> 1;              // row
  int x0   = (rem & 1) << 5;        // 32-pixel segment base
  int tid  = threadIdx.x;

  const bf16* xsrc = x16 + b * (CIN * PIX);
  const bf16* hsrc = h16 + b * (HID * PIX);

  // Cooperative im2col gather straight into B-fragment layout:
  // ISA 16-bit B 32x16 layout: lane<16 -> N=lane, K_local = (e<8 ? e : e+8)
  //                            lane>=16 -> N=lane-16, K_local += 8
  // LDS flat index = (((kc*2)+tileN)*32 + lane)*16 + e
  for (int idx = tid; idx < KCH * 2 * 32 * 16; idx += 256) {
    int e     = idx & 15;
    int lane  = (idx >> 4) & 31;
    int tileN = (idx >> 9) & 1;
    int kc    = idx >> 10;
    int kl    = (e < 8 ? e : e + 8) + ((lane < 16) ? 0 : 8);
    int K     = kc * 32 + kl;
    int ch    = K / 9;
    int tap   = K - ch * 9;
    int ty    = tap / 3;
    int dy    = ty - 1;
    int dx    = (tap - ty * 3) - 1;
    int n     = (lane & 15) + (tileN << 4);
    int xx    = x0 + n + dx;
    int yy    = y0 + dy;
    float v   = 0.f;
    if (xx >= 0 && xx < WW && yy >= 0 && yy < HH) {
      int p = yy * WW + xx;
      v = (ch < CIN) ? (float)xsrc[ch * PIX + p]
                     : (float)hsrc[(ch - CIN) * PIX + p];
    }
    ldsBs[idx] = (bf16)v;
  }
  __syncthreads();

  int wave = tid >> 5;
  int lane = tid & 31;
  int mt0  = wave * 2;              // two 16-channel tiles per wave -> 32 channels

  const v16bf* pw = (const v16bf*)wp;
  v8f acc00 = {}, acc01 = {}, acc10 = {}, acc11 = {};
#pragma unroll
  for (int kc = 0; kc < KCH; ++kc) {
    v16bf b0 = ldsB[kc * 64 + lane];
    v16bf b1 = ldsB[kc * 64 + 32 + lane];
    v16bf a0 = pw[(mt0 * KCH + kc) * 32 + lane];
    v16bf a1 = pw[((mt0 + 1) * KCH + kc) * 32 + lane];
    acc00 = __builtin_amdgcn_wmma_f32_16x16x32_bf16(false, a0, false, b0,
                                                    (short)0, acc00, false, false);
    acc10 = __builtin_amdgcn_wmma_f32_16x16x32_bf16(false, a1, false, b0,
                                                    (short)0, acc10, false, false);
    acc01 = __builtin_amdgcn_wmma_f32_16x16x32_bf16(false, a0, false, b1,
                                                    (short)0, acc01, false, false);
    acc11 = __builtin_amdgcn_wmma_f32_16x16x32_bf16(false, a1, false, b1,
                                                    (short)0, acc11, false, false);
  }

  // C/D layout: VGPR r, lanes 0-15: M=r, N=lane ; lanes 16-31: M=r+8, N=lane-16
  int m_off = (lane < 16) ? 0 : 8;
  int n     = lane & 15;
  int pix0  = y0 * WW + x0 + n;     // N-tile 0
  float* gout = gates + b * (GOUT * PIX);
#pragma unroll
  for (int r = 0; r < 8; ++r) {
    int row0 = (mt0 * 16 + r + m_off) * PIX;
    int row1 = ((mt0 + 1) * 16 + r + m_off) * PIX;
    gout[row0 + pix0]      = acc00[r];
    gout[row0 + pix0 + 16] = acc01[r];
    gout[row1 + pix0]      = acc10[r];
    gout[row1 + pix0 + 16] = acc11[r];
  }
}

// ---------------------------------------------------------------- pointwise LSTM cell
__global__ __launch_bounds__(256) void lstm_point_kernel(
    const float* __restrict__ gates,   // [BB,GOUT,HH,WW]
    const float* __restrict__ bx, const float* __restrict__ bh,
    float* __restrict__ c, bf16* __restrict__ h16,
    float* __restrict__ hout) {        // d_out + t*BB*HID*PIX, layout [BB,HID,HH,WW]
  int i   = blockIdx.x * 256 + threadIdx.x;   // < BB*HID*PIX = 1048576
  int pix = i & (PIX - 1);
  int hid = (i >> 12) & (HID - 1);
  int b   = i >> 18;
  const float* g = gates + b * (GOUT * PIX) + pix;
  float gi = g[hid * PIX]         + bx[hid]       + bh[hid];
  float gf = g[(HID + hid) * PIX] + bx[HID + hid] + bh[HID + hid];
  float gg = g[(2*HID + hid)*PIX] + bx[2*HID+hid] + bh[2*HID+hid];
  float go = g[(3*HID + hid)*PIX] + bx[3*HID+hid] + bh[3*HID+hid];
  float iv = 1.f / (1.f + __expf(-gi));
  float fv = 1.f / (1.f + __expf(-gf));
  float ov = 1.f / (1.f + __expf(-go));
  float gv = tanhf(gg);
  float cv = fv * c[i] + iv * gv;
  c[i] = cv;
  float hv = ov * tanhf(cv);
  hout[i] = hv;
  h16[i] = (bf16)hv;
}

// ---------------------------------------------------------------- launch
extern "C" void kernel_launch(void* const* d_in, const int* in_sizes, int n_in,
                              void* d_out, int out_size, void* d_ws, size_t ws_size,
                              hipStream_t stream) {
  const float* x     = (const float*)d_in[0];
  const float* w_x2h = (const float*)d_in[1];
  const float* b_x2h = (const float*)d_in[2];
  const float* w_h2h = (const float*)d_in[3];
  const float* b_h2h = (const float*)d_in[4];
  float* out = (float*)d_out;
  char*  ws  = (char*)d_ws;

  const size_t SZ_X16   = (size_t)TT * BB * CIN * PIX * sizeof(bf16);   // 16,777,216
  const size_t SZ_WP    = (size_t)16 * KCH * 32 * 16 * sizeof(bf16);    //    442,368
  const size_t SZ_H16   = (size_t)BB * HID * PIX * sizeof(bf16);        //  2,097,152
  const size_t SZ_C     = (size_t)BB * HID * PIX * sizeof(float);       //  4,194,304

  bf16*  x16   = (bf16*)ws;
  bf16*  wp    = (bf16*)(ws + SZ_X16);
  bf16*  h16   = (bf16*)(ws + SZ_X16 + SZ_WP);
  float* cst   = (float*)(ws + SZ_X16 + SZ_WP + SZ_H16);
  float* gates = (float*)(ws + SZ_X16 + SZ_WP + SZ_H16 + SZ_C);

  int nX = TT * BB * CIN * PIX;                       // 8,388,608
  cvt_x_kernel<<<(nX + 255) / 256, 256, 0, stream>>>(x, x16, nX);
  pack_w_kernel<<<(16 * KCH * 32 * 16) / 256, 256, 0, stream>>>(w_x2h, w_h2h, wp);
  zero_state_kernel<<<(BB * HID * PIX) / 256, 256, 0, stream>>>(cst, h16);

  for (int t = 0; t < TT; ++t) {
    conv_gates_kernel<<<BB * HH * (WW / 32), 256, 0, stream>>>(
        x16 + (size_t)t * BB * CIN * PIX, h16, wp, gates);
    lstm_point_kernel<<<(BB * HID * PIX) / 256, 256, 0, stream>>>(
        gates, b_x2h, b_h2h, cst, h16, out + (size_t)t * BB * HID * PIX);
  }
}
